// PricePredictionModel_51659866636566
// MI455X (gfx1250) — matrix operand order repaired and checked by reading, not compiled, not run on hardware
//
#include <hip/hip_runtime.h>
#include <hip/hip_bf16.h>

// ---------------------------------------------------------------------------
// Persistent fused 2-layer LSTM encoder (T=512) + 2-layer LSTM decoder (64) +
// linear head.  One workgroup (512 thr = 16 waves) owns 16 batch rows and
// runs the ENTIRE sequence; batch rows are independent -> no inter-WG sync.
// Each wave owns TWO 16-wide gate-column tiles; the 3 active 512x128 weight
// matrices live in registers as bf16 WMMA B-fragments.  Elementwise cell:
// thread (w,lane) owns row w, hidden lane*4..lane*4+3  -> vectorized LDS I/O.
// ---------------------------------------------------------------------------

#define B_TOT 2048
#define T_SEQ 512
#define H_DIM 128
#define G_DIM 512   // 4*H
#define OUT_N 64
#define MB    16    // batch rows per workgroup
#define NTHR  512   // 16 waves

typedef __attribute__((ext_vector_type(16))) __bf16 v16bf;
typedef __attribute__((ext_vector_type(8)))  float  v8f;

union BFrag {
    v16bf v;
    unsigned short s[16];
    uint4 q[2];
};
union F4 { float4 v; float f[4]; };

__device__ __forceinline__ unsigned short f2bf(float f) {
    unsigned u = __float_as_uint(f);
    unsigned r = (u + 0x7FFFu + ((u >> 16) & 1u)) >> 16;   // round-to-nearest-even
    return (unsigned short)r;
}
// Branch-free activations: v_exp_f32 + v_rcp_f32, no OCML calls.
__device__ __forceinline__ float fsigm(float x) {
    return __builtin_amdgcn_rcpf(1.f + __expf(-x));
}
__device__ __forceinline__ float ftanh(float x) {
    return 2.f * __builtin_amdgcn_rcpf(1.f + __expf(-2.f * x)) - 1.f;
}

// B operand (gates = act @ W^T, so B[k][n] = W[n][k]) for one column tile,
// fp32 -> bf16 into this file's contiguous-K B order: lane L holds N = ct*16+(L&15);
// 16 consecutive K per lane starting at kb + (L/16)*16.
__device__ __forceinline__ void load_wfrags(BFrag Wf[4], const float* __restrict__ W,
                                            int ct, int lane) {
    int n = ct * 16 + (lane & 15);
#pragma unroll
    for (int k = 0; k < 4; k++) {
        const float* p = W + n * H_DIM + k * 32 + ((lane >> 4) << 4);
#pragma unroll
        for (int j = 0; j < 16; j++) Wf[k].s[j] = f2bf(p[j]);
    }
}

// acc{0,1} += hb(16xH bf16 row-major LDS) @ W^T for this wave's two tiles.
// bf16 A layout: lane L holds M = L&15; VGPRs 0-7: K=kb+(L/16)*16+0..15
// (contiguous, matching B)  -> two ds_load_b128, shared by 2 WMMAs.
__device__ __forceinline__ void mm_acc2(v8f& acc0, v8f& acc1,
                                        const unsigned short* hb, int aoff,
                                        const BFrag W0[4], const BFrag W1[4]) {
#pragma unroll
    for (int k = 0; k < 4; k++) {
        BFrag a;
        a.q[0] = *reinterpret_cast<const uint4*>(hb + aoff + k * 32);
        a.q[1] = *reinterpret_cast<const uint4*>(hb + aoff + k * 32 + 8);
        acc0 = __builtin_amdgcn_wmma_f32_16x16x32_bf16(
            false, a.v, false, W0[k].v, (short)0, acc0, false, false);
        acc1 = __builtin_amdgcn_wmma_f32_16x16x32_bf16(
            false, a.v, false, W1[k].v, (short)0, acc1, false, false);
    }
}

// f32 16x16 C layout: lane L -> col N = L&15 (+tile base); VGPR v -> row
// M = (L/16)*8 + v.
__device__ __forceinline__ void store_gates(v8f acc, float* gates, int ct, int lane) {
    int col = ct * 16 + (lane & 15);
    int rb  = (lane >> 4) << 3;
#pragma unroll
    for (int v = 0; v < 8; v++) gates[(rb + v) * G_DIM + col] = acc[v];
}

// Elementwise LSTM cell: thread (w,lane) owns row w, hid = lane*4..+3.
// Gate blocks (i,f,g,o) are 128 floats = 32 float4s apart -> b128 loads off
// one base pointer with immediate offsets.  h written as one packed b64.
// Returns the fp32 h values in hout[] (used by the linear head).
template <bool HASX>
__device__ __forceinline__ void cell_update(const float* gates, const float* bias,
                                            const float* wxv, const float* xs,
                                            unsigned short* hbuf, float creg[4],
                                            int w, int lane, float hout[4]) {
    const int hid = lane * 4;
    const float4* gp = reinterpret_cast<const float4*>(gates + w * G_DIM + hid);
    const float4* bp = reinterpret_cast<const float4*>(bias + hid);
    F4 vi, vf, vg, vo, bi, bf4, bg4, bo4;
    vi.v = gp[0];  vf.v = gp[32];  vg.v = gp[64];  vo.v = gp[96];
    bi.v = bp[0];  bf4.v = bp[32]; bg4.v = bp[64]; bo4.v = bp[96];
    float xi = 0.f;
    F4 wi, wf4, wg4, wo4;
    if (HASX) {
        xi = xs[w];
        const float4* wp = reinterpret_cast<const float4*>(wxv + hid);
        wi.v = wp[0]; wf4.v = wp[32]; wg4.v = wp[64]; wo4.v = wp[96];
    }
    unsigned short hs[4];
#pragma unroll
    for (int j = 0; j < 4; j++) {
        float gi = vi.f[j] + bi.f[j];
        float gf = vf.f[j] + bf4.f[j];
        float gg = vg.f[j] + bg4.f[j];
        float go = vo.f[j] + bo4.f[j];
        if (HASX) {
            gi += xi * wi.f[j];
            gf += xi * wf4.f[j];
            gg += xi * wg4.f[j];
            go += xi * wo4.f[j];
        }
        float c = fsigm(gf) * creg[j] + fsigm(gi) * ftanh(gg);
        creg[j] = c;
        float h = fsigm(go) * ftanh(c);
        hout[j] = h;
        hs[j] = f2bf(h);
    }
    uint2 pk;
    pk.x = (unsigned)hs[0] | ((unsigned)hs[1] << 16);
    pk.y = (unsigned)hs[2] | ((unsigned)hs[3] << 16);
    *reinterpret_cast<uint2*>(hbuf + w * H_DIM + hid) = pk;
}

__global__ __launch_bounds__(NTHR, 1) void lstm_seq2seq_kernel(
    const float* __restrict__ x,
    const float* __restrict__ eWih0, const float* __restrict__ eWhh0,
    const float* __restrict__ ebih0, const float* __restrict__ ebhh0,
    const float* __restrict__ eWih1, const float* __restrict__ eWhh1,
    const float* __restrict__ ebih1, const float* __restrict__ ebhh1,
    const float* __restrict__ dWih0, const float* __restrict__ dWhh0,
    const float* __restrict__ dbih0, const float* __restrict__ dbhh0,
    const float* __restrict__ dWih1, const float* __restrict__ dWhh1,
    const float* __restrict__ dbih1, const float* __restrict__ dbhh1,
    const float* __restrict__ linW, const float* __restrict__ linb,
    float* __restrict__ out)
{
    __shared__ float gates[MB * G_DIM];                        // 32 KB fp32
    __shared__ __align__(16) unsigned short h0buf[MB * H_DIM]; // 4 KB bf16
    __shared__ __align__(16) unsigned short h1buf[MB * H_DIM]; // 4 KB bf16
    __shared__ __align__(16) float bias0[G_DIM];               // 2 KB each
    __shared__ __align__(16) float bias1[G_DIM];
    __shared__ __align__(16) float wxv[G_DIM];
    __shared__ __align__(16) float linw_s[H_DIM];
    __shared__ float xbuf[MB];
    __shared__ float linb_s;

    const int tid  = threadIdx.x;
    const int lane = tid & 31;
    const int w    = tid >> 5;            // wave id, 0..15 (== cell row)
    const int ct0  = 2 * w;               // this wave's two gate-column tiles
    const int ct1  = 2 * w + 1;
    const int row0 = blockIdx.x * MB;     // global batch row base
    // per-lane A-fragment LDS element offset (same for every matmul)
    const int aoff = (lane & 15) * H_DIM + ((lane >> 4) << 4);

    // -------- encoder phase: weights -> registers (bf16 B fragments) --------
    BFrag Wh0[2][4], Wx1[2][4], Wh1[2][4];
    load_wfrags(Wh0[0], eWhh0, ct0, lane);  load_wfrags(Wh0[1], eWhh0, ct1, lane);
    load_wfrags(Wx1[0], eWih1, ct0, lane);  load_wfrags(Wx1[1], eWih1, ct1, lane);
    load_wfrags(Wh1[0], eWhh1, ct0, lane);  load_wfrags(Wh1[1], eWhh1, ct1, lane);

    for (int i = tid; i < G_DIM; i += NTHR) {
        bias0[i] = ebih0[i] + ebhh0[i];
        bias1[i] = ebih1[i] + ebhh1[i];
        wxv[i]   = eWih0[i];              // (4H,1) input weights, rank-1 term
    }
    if (tid < H_DIM) linw_s[tid] = linW[tid];
    if (tid == 0)    linb_s = linb[0];
    for (int i = tid; i < MB * H_DIM; i += NTHR) { h0buf[i] = 0; h1buf[i] = 0; }
    float c0[4] = {0.f, 0.f, 0.f, 0.f}, c1[4] = {0.f, 0.f, 0.f, 0.f};
    float hout[4];
    __syncthreads();

    // ------------------------------ encoder ------------------------------
    for (int t = 0; t < T_SEQ; t++) {
        if (tid < MB) {
            xbuf[tid] = x[(row0 + tid) * T_SEQ + t];
            int tp = (t + 16 < T_SEQ) ? t + 16 : T_SEQ - 1;
            __builtin_prefetch(&x[(row0 + tid) * T_SEQ + tp], 0, 1);
        }
        __syncthreads();
        { // layer0: gates = h0 @ Whh0^T   (x term folded into cell_update)
            v8f a0 = {}, a1 = {};
            mm_acc2(a0, a1, h0buf, aoff, Wh0[0], Wh0[1]);
            store_gates(a0, gates, ct0, lane);
            store_gates(a1, gates, ct1, lane);
        }
        __syncthreads();
        cell_update<true>(gates, bias0, wxv, xbuf, h0buf, c0, w, lane, hout);
        __syncthreads();
        { // layer1: gates = h0_new @ Wih1^T + h1 @ Whh1^T
            v8f a0 = {}, a1 = {};
            mm_acc2(a0, a1, h0buf, aoff, Wx1[0], Wx1[1]);
            mm_acc2(a0, a1, h1buf, aoff, Wh1[0], Wh1[1]);
            store_gates(a0, gates, ct0, lane);
            store_gates(a1, gates, ct1, lane);
        }
        __syncthreads();
        cell_update<false>(gates, bias1, nullptr, nullptr, h1buf, c1, w, lane, hout);
        __syncthreads();
    }

    // --------------- swap register weights to decoder phase ---------------
    load_wfrags(Wh0[0], dWhh0, ct0, lane);  load_wfrags(Wh0[1], dWhh0, ct1, lane);
    load_wfrags(Wx1[0], dWih1, ct0, lane);  load_wfrags(Wx1[1], dWih1, ct1, lane);
    load_wfrags(Wh1[0], dWhh1, ct0, lane);  load_wfrags(Wh1[1], dWhh1, ct1, lane);
    for (int i = tid; i < G_DIM; i += NTHR) {
        bias0[i] = dbih0[i] + dbhh0[i];
        bias1[i] = dbih1[i] + dbhh1[i];
        wxv[i]   = dWih0[i];
    }
    if (tid < MB) xbuf[tid] = x[(row0 + tid) * T_SEQ + (T_SEQ - 1)]; // inp0 = x[:,-1]
    __syncthreads();

    // ------------------------------ decoder ------------------------------
    for (int t = 0; t < OUT_N; t++) {
        {
            v8f a0 = {}, a1 = {};
            mm_acc2(a0, a1, h0buf, aoff, Wh0[0], Wh0[1]);
            store_gates(a0, gates, ct0, lane);
            store_gates(a1, gates, ct1, lane);
        }
        __syncthreads();
        cell_update<true>(gates, bias0, wxv, xbuf, h0buf, c0, w, lane, hout);
        __syncthreads();
        {
            v8f a0 = {}, a1 = {};
            mm_acc2(a0, a1, h0buf, aoff, Wx1[0], Wx1[1]);
            mm_acc2(a0, a1, h1buf, aoff, Wh1[0], Wh1[1]);
            store_gates(a0, gates, ct0, lane);
            store_gates(a1, gates, ct1, lane);
        }
        __syncthreads();
        cell_update<false>(gates, bias1, nullptr, nullptr, h1buf, c1, w, lane, hout);
        // linear head: this thread just produced h1[w][lane*4..+3] in hout.
        {
            const int hid = lane * 4;
            float part = hout[0] * linw_s[hid]     + hout[1] * linw_s[hid + 1]
                       + hout[2] * linw_s[hid + 2] + hout[3] * linw_s[hid + 3];
#pragma unroll
            for (int off = 16; off; off >>= 1) part += __shfl_xor(part, off, 32);
            if (lane == 0) {
                float pred = part + linb_s;
                xbuf[w] = pred;                       // feedback input (wave-local)
                out[(row0 + w) * OUT_N + t] = pred;   // preds.T -> (B, OUT)
            }
        }
        __syncthreads();
    }
}

extern "C" void kernel_launch(void* const* d_in, const int* in_sizes, int n_in,
                              void* d_out, int out_size, void* d_ws, size_t ws_size,
                              hipStream_t stream) {
    (void)in_sizes; (void)n_in; (void)out_size; (void)d_ws; (void)ws_size;
    const float* p[19];
    for (int i = 0; i < 19; i++) p[i] = (const float*)d_in[i];
    hipLaunchKernelGGL(lstm_seq2seq_kernel, dim3(B_TOT / MB), dim3(NTHR), 0, stream,
                       p[0], p[1], p[2], p[3], p[4], p[5], p[6], p[7], p[8],
                       p[9], p[10], p[11], p[12], p[13], p[14], p[15], p[16],
                       p[17], p[18], (float*)d_out);
}
